// C4ByteNibbleVM_62380105007523
// MI455X (gfx1250) — compile-verified, matrix operand order checked by reading, not run on hardware
//
#include <hip/hip_runtime.h>

// CDNA5 / gfx1250, wave32.
// Exact-integer reformulation of the one-hot "neural VM":
//   out_byte = ((a + b) mod 2^32) ^ a  per byte, re-encoded as one-hot f32.
// One wave handles 2 words: 16 byte-rows (8 of a, 8 of b) are decoded
// one-hot -> index with V_WMMA_F32_16X16X32_F16 (A = one-hot, B = iota),
// integer math happens on SALU after v_readlane, and the one-hot output is
// written with non-temporal B128 stores. Two WMMA accumulators break the
// D->C RAW chain; iota-B per chunk is a packed-f16 add of a constant splat.

typedef __attribute__((ext_vector_type(16))) _Float16 v16h;
typedef __attribute__((ext_vector_type(8)))  float    v8f;
typedef __attribute__((ext_vector_type(4)))  float    v4f;

__global__ __launch_bounds__(256, 1) void c4vm_wmma_kernel(
    const float* __restrict__ Ain,   // [nwords,4,256] one-hot f32
    const float* __restrict__ Bin,   // [nwords,4,256] one-hot f32
    float* __restrict__ Out,         // [nwords,4,256] one-hot f32
    int nwords)
{
  const int lane = threadIdx.x & 31;
  const int wave = blockIdx.x * (blockDim.x >> 5) + (threadIdx.x >> 5);
  const int w0   = wave * 2;                    // this wave's first word
  if (w0 >= nwords) return;                     // wave-uniform guard

  // ---- A-operand addressing (ISA 16-bit 16x32 A layout) -------------------
  // lane L -> row M = L%16 ; lanes 0-15 hold K in {0..7,16..23},
  // lanes 16-31 hold K in {8..15,24..31} (per 32-wide K chunk).
  const int M  = lane & 15;
  const int kb = (lane >> 4) << 3;              // 0 or 8
  const float* rowp = (M < 8)
      ? (Ain + ((size_t)(w0 * 4 + M)       << 8))   // rows 0-7: a bytes (w0,w1)
      : (Bin + ((size_t)(w0 * 4 + (M - 8)) << 8));  // rows 8-15: b bytes

  // ---- B-operand: iota over K for chunk 0 (all 16 columns identical) ------
  // bi[i] = f16(kb + i), bi[8+i] = f16(kb + 16 + i); kb in {0,8}, so all
  // values and later +32*c offsets are exact in f16.
  const _Float16 kbh = (_Float16)(float)kb;
  v16h biota;
#pragma unroll
  for (int i = 0; i < 8; ++i) {
    biota[i]     = kbh + (_Float16)(float)i;
    biota[i + 8] = kbh + (_Float16)(float)(16 + i);
  }

  // ---- decode: 8 chunks of 32 columns, two interleaved accumulators -------
  v8f acc0 = {};
  v8f acc1 = {};
#pragma unroll
  for (int c = 0; c < 8; ++c) {
    const float* p = rowp + c * 32 + kb;
    v4f q0 = __builtin_nontemporal_load((const v4f*)(p + 0));
    v4f q1 = __builtin_nontemporal_load((const v4f*)(p + 4));
    v4f q2 = __builtin_nontemporal_load((const v4f*)(p + 16));
    v4f q3 = __builtin_nontemporal_load((const v4f*)(p + 20));
    v16h a;
#pragma unroll
    for (int i = 0; i < 4; ++i) {
      a[i]      = (_Float16)q0[i];
      a[i + 4]  = (_Float16)q1[i];
      a[i + 8]  = (_Float16)q2[i];
      a[i + 12] = (_Float16)q3[i];
    }
    // bi = biota + splat(32*c): whole-vector packed-f16 add (v_pk_add_f16),
    // computed from the chunk-0 base so there is no serial dependency chain.
    v16h bi;
    if (c == 0) {
      bi = biota;
    } else {
      const _Float16 off = (_Float16)(float)(c * 32);
      v16h offv;
#pragma unroll
      for (int i = 0; i < 16; ++i) offv[i] = off;
      bi = biota + offv;
    }
    // D = A x B + C : v_wmma_f32_16x16x32_f16 (even/odd accumulators)
    if ((c & 1) == 0)
      acc0 = __builtin_amdgcn_wmma_f32_16x16x32_f16(
          false, a, false, bi, (short)0, acc0, false, false);
    else
      acc1 = __builtin_amdgcn_wmma_f32_16x16x32_f16(
          false, a, false, bi, (short)0, acc1, false, false);
  }

  // ---- extract byte indices (C/D layout: VGPR r = row r / r+8) ------------
  int dv[8];
#pragma unroll
  for (int r = 0; r < 8; ++r) dv[r] = (int)(acc0[r] + acc1[r]); // exact 0..255

  unsigned ra[8], rb[8];
#pragma unroll
  for (int r = 0; r < 8; ++r) {
    ra[r] = (unsigned)__builtin_amdgcn_readlane(dv[r], 0);   // rows 0-7  (a)
    rb[r] = (unsigned)__builtin_amdgcn_readlane(dv[r], 16);  // rows 8-15 (b)
  }

  // ---- the whole VM on SALU: s = a+b (mod 2^32), out = s ^ a --------------
  const unsigned wa0 = ra[0] | (ra[1] << 8) | (ra[2] << 16) | (ra[3] << 24);
  const unsigned wa1 = ra[4] | (ra[5] << 8) | (ra[6] << 16) | (ra[7] << 24);
  const unsigned wb0 = rb[0] | (rb[1] << 8) | (rb[2] << 16) | (rb[3] << 24);
  const unsigned wb1 = rb[4] | (rb[5] << 8) | (rb[6] << 16) | (rb[7] << 24);
  const unsigned o0 = (wa0 + wb0) ^ wa0;        // word w0 output bytes
  const unsigned o1 = (wa1 + wb1) ^ wa1;        // word w1 output bytes

  // ---- one-hot output: 2 words * 1024 f32; lane owns 64 floats ------------
  const unsigned sel = (lane < 16) ? o0 : o1;               // rows 0-3 / 4-7
  const unsigned hot = (sel >> (((lane >> 2) & 3) * 8)) & 0xFFu;

  float* base = Out + ((size_t)w0 << 10);
  float* myz  = base + lane * 64;
  const v4f z = {0.f, 0.f, 0.f, 0.f};
#pragma unroll
  for (int t = 0; t < 16; ++t)
    __builtin_nontemporal_store(z, (v4f*)(myz + t * 4));

  // hot column lies inside exactly one lane's 64-float span; same-wave stores
  // to the same address complete in order, so this lands after the zeros.
  const unsigned colbase = (lane & 3) * 64;
  if (hot - colbase < 64u)
    __builtin_nontemporal_store(1.0f, base + ((lane >> 2) << 8) + hot);
}

extern "C" void kernel_launch(void* const* d_in, const int* in_sizes, int n_in,
                              void* d_out, int out_size, void* d_ws, size_t ws_size,
                              hipStream_t stream) {
  const float* a = (const float*)d_in[0];   // a_bytes [B,4,256] f32
  const float* b = (const float*)d_in[1];   // b_bytes [B,4,256] f32
  float* out = (float*)d_out;               // [B,4,256] f32

  const int nwords = in_sizes[0] / 1024;    // B
  const int pairs  = (nwords + 1) / 2;      // 2 words per wave
  const int wavesPerBlock = 8;              // 256 threads, wave32
  const int blocks = (pairs + wavesPerBlock - 1) / wavesPerBlock;

  c4vm_wmma_kernel<<<blocks, 256, 0, stream>>>(a, b, out, nwords);
}
